// Augmenter_separate_8005819040426
// MI455X (gfx1250) — compile-verified, matrix operand order checked by reading, not run on hardware
//
#include <hip/hip_runtime.h>
#include <hip/hip_bf16.h>

typedef __attribute__((ext_vector_type(16))) _Float16 v16h;
typedef __attribute__((ext_vector_type(8)))  _Float16 v8h;
typedef __attribute__((ext_vector_type(8)))  float    v8f;

#define KFAC   4
#define DFAC   64
#define HDIM   128
#define NITEMS 8192
#define BATCH  64

// ---------------- prep kernels ----------------

// zcb[kb][h] = b1[h] + sum_d z[kb][d] * W1[d][h]   (kb = k*64+b), stored f16
__global__ void prep_zcb(const float* __restrict__ z, const float* __restrict__ W1,
                         const float* __restrict__ b1, _Float16* __restrict__ zcbp) {
  int gid = blockIdx.x * 256 + threadIdx.x;           // 32768
  int kb = gid >> 7, h = gid & 127;
  float acc = b1[h];
  #pragma unroll 8
  for (int d = 0; d < DFAC; ++d) acc += z[kb * DFAC + d] * W1[d * HDIM + h];
  zcbp[gid] = (_Float16)acc;
}

// ic[n][h] = sum_d items[n][d] * W1[64+d][h], stored f16
__global__ void prep_ic(const float* __restrict__ items, const float* __restrict__ W1,
                        _Float16* __restrict__ icp) {
  int gid = blockIdx.x * 256 + threadIdx.x;           // 1048576
  int n = gid >> 7, h = gid & 127;
  float acc = 0.f;
  #pragma unroll 8
  for (int d = 0; d < DFAC; ++d) acc += items[n * DFAC + d] * W1[(DFAC + d) * HDIM + h];
  icp[gid] = (_Float16)acc;
}

// Pack W2 (K=128 x N=128, row-major f32) into WMMA B-fragment order, f16.
// Layout: w2p[((j*4+c)*32 + lane)*16 + t], lane holds column N = 16j+(lane&15),
// K = 32c + (t<8 ? 8*(lane>>4)+t : 16 + 8*(lane>>4)+(t-8))
__global__ void pack_w2(const float* __restrict__ W2, _Float16* __restrict__ w2p) {
  int tid = blockIdx.x * 256 + threadIdx.x;           // 16384
  int t = tid & 15, lane = (tid >> 4) & 31, c = (tid >> 9) & 3, j = tid >> 11;
  int half = lane >> 4;
  int N = 16 * j + (lane & 15);
  int K = 32 * c + ((t < 8) ? (8 * half + t) : (16 + 8 * half + (t - 8)));
  w2p[tid] = (_Float16)W2[K * HDIM + N];
}

__global__ void zero_scalar(float* __restrict__ out) {
  if (threadIdx.x == 0) out[(size_t)BATCH * NITEMS] = 0.f;
}

// ---------------- main fused kernel ----------------
__global__ __launch_bounds__(256) void aug_main(
    const _Float16* __restrict__ icp, const _Float16* __restrict__ zcbp,
    const _Float16* __restrict__ w2p,
    const float* __restrict__ cates, const int* __restrict__ x,
    const float* __restrict__ delta, const float* __restrict__ b2,
    const float* __restrict__ W3, const float* __restrict__ b3,
    float* __restrict__ out) {
  __shared__ _Float16 sW2[16384];       // 32 KB packed W2 B-fragments
  __shared__ float    sRed[8][16][17];  // per-wave column reduction (padded)
  __shared__ float    sReg[8];

  const int tid = threadIdx.x, wave = tid >> 5, lane = tid & 31;
  const int half = lane >> 4, col = lane & 15;
  const int n0 = blockIdx.x * 16;
  const int b = blockIdx.y * 8 + wave;

  // cooperative load of packed W2 into LDS (uint4 = 8 halfs per load)
  {
    const uint4* src = (const uint4*)w2p;
    uint4* dst = (uint4*)sW2;
    for (int i = tid; i < 2048; i += 256) dst[i] = src[i];
  }
  __syncthreads();

  // per-lane constants: b2 / W3 for this lane's output column in each j-chunk
  float b2a[8], w3a[8];
  #pragma unroll
  for (int j = 0; j < 8; ++j) {
    b2a[j] = b2[16 * j + col];
    w3a[j] = W3[16 * j + col];
  }
  const float b3v = b3[0];

  // ic A-fragments for this n-tile (reused across all 4 k)
  v8h iclo[4], ichi[4];
  #pragma unroll
  for (int c = 0; c < 4; ++c) {
    const _Float16* base = icp + (size_t)(n0 + col) * HDIM + 32 * c + 8 * half;
    iclo[c] = *(const v8h*)(base);
    ichi[c] = *(const v8h*)(base + 16);
  }

  float outacc = 0.f, regacc = 0.f;

  for (int k = 0; k < KFAC; ++k) {
    const int kb = k * BATCH + b;

    // A = relu(ic + zcb[k,b])  (16 rows x 128, f16 WMMA A-fragments)
    v16h A[4];
    #pragma unroll
    for (int c = 0; c < 4; ++c) {
      const _Float16* zb = zcbp + (size_t)kb * HDIM + 32 * c + 8 * half;
      v8h lo = iclo[c] + *(const v8h*)zb;
      v8h hi = ichi[c] + *(const v8h*)(zb + 16);
      #pragma unroll
      for (int t = 0; t < 8; ++t) {
        lo[t] = lo[t] > (_Float16)0 ? lo[t] : (_Float16)0;
        hi[t] = hi[t] > (_Float16)0 ? hi[t] : (_Float16)0;
      }
      A[c] = __builtin_shufflevector(lo, hi, 0, 1, 2, 3, 4, 5, 6, 7,
                                     8, 9, 10, 11, 12, 13, 14, 15);
    }

    // h2 = relu(A @ W2 + b2); fold W3 dot into the accumulator registers
    float partial[8] = {0, 0, 0, 0, 0, 0, 0, 0};
    #pragma unroll
    for (int j = 0; j < 8; ++j) {
      v8f acc;
      #pragma unroll
      for (int i = 0; i < 8; ++i) acc[i] = b2a[j];
      #pragma unroll
      for (int c = 0; c < 4; ++c) {
        v16h B = *(const v16h*)(sW2 + ((j * 4 + c) * 32 + lane) * 16);
        acc = __builtin_amdgcn_wmma_f32_16x16x32_f16(false, A[c], false, B,
                                                     (short)0, acc, false, false);
      }
      #pragma unroll
      for (int i = 0; i < 8; ++i) {
        float h2 = acc[i] > 0.f ? acc[i] : 0.f;
        partial[i] += h2 * w3a[j];
      }
    }

    // reduce over the 16 H-columns held across lanes (per-wave LDS; DS is
    // in-order within a wave so only a compiler barrier is needed)
    #pragma unroll
    for (int i = 0; i < 8; ++i) sRed[wave][i + 8 * half][col] = partial[i];
    asm volatile("" ::: "memory");
    if (lane < 16) {
      float wsum = 0.f;
      #pragma unroll
      for (int c2 = 0; c2 < 16; ++c2) wsum += sRed[wave][lane][c2];
      const int n = n0 + lane;
      float w = (wsum + b3v) * cates[n * KFAC + k];
      float dl = delta[(size_t)kb * NITEMS + n];
      float lg = __logf(dl) - __logf(1.f - dl) + w;   // log(d) - log1p(-d)
      float p = 1.f / (1.f + __expf(lg * -5.0f));     // sigmoid(lg / 0.2)
      regacc += p;                                    // |p| == p
      outacc += (x[(size_t)b * NITEMS + n] == 1) ? (1.f - p) : p;
    }
    asm volatile("" ::: "memory");
  }

  if (lane < 16) out[(size_t)b * NITEMS + n0 + lane] = outacc * 0.25f;

  // reg-loss reduction: wave -> block -> one global atomic per block
  #pragma unroll
  for (int off = 16; off >= 1; off >>= 1) regacc += __shfl_xor(regacc, off, 32);
  if (lane == 0) sReg[wave] = regacc;
  __syncthreads();
  if (tid == 0) {
    float tot = 0.f;
    #pragma unroll
    for (int wv = 0; wv < 8; ++wv) tot += sReg[wv];
    atomicAdd(out + (size_t)BATCH * NITEMS,
              tot * (1.0f / ((float)BATCH * NITEMS * (KFAC - 1))));
  }
}

// ---------------- launcher ----------------
extern "C" void kernel_launch(void* const* d_in, const int* in_sizes, int n_in,
                              void* d_out, int out_size, void* d_ws, size_t ws_size,
                              hipStream_t stream) {
  const float* cates = (const float*)d_in[0];
  // d_in[1] = h (unused by forward)
  const float* z     = (const float*)d_in[2];
  const int*   x     = (const int*)d_in[3];
  const float* delta = (const float*)d_in[4];
  const float* items = (const float*)d_in[5];
  const float* W1    = (const float*)d_in[6];
  const float* b1    = (const float*)d_in[7];
  const float* W2    = (const float*)d_in[8];
  const float* b2    = (const float*)d_in[9];
  const float* W3    = (const float*)d_in[10];
  const float* b3    = (const float*)d_in[11];
  float* out = (float*)d_out;

  // workspace layout (f16): ic 8192x128 (2 MB) | zcb 256x128 (64 KB) | w2pack 16384 (32 KB)
  _Float16* icp  = (_Float16*)d_ws;
  _Float16* zcbp = (_Float16*)((char*)d_ws + (size_t)NITEMS * HDIM * 2);
  _Float16* w2p  = (_Float16*)((char*)d_ws + (size_t)NITEMS * HDIM * 2 +
                               (size_t)KFAC * BATCH * HDIM * 2);

  prep_zcb<<<dim3((KFAC * BATCH * HDIM) / 256), dim3(256), 0, stream>>>(z, W1, b1, zcbp);
  prep_ic<<<dim3((NITEMS * HDIM) / 256), dim3(256), 0, stream>>>(items, W1, icp);
  pack_w2<<<dim3((HDIM * HDIM) / 256), dim3(256), 0, stream>>>(W2, w2p);
  zero_scalar<<<dim3(1), dim3(64), 0, stream>>>(out);

  aug_main<<<dim3(NITEMS / 16, BATCH / 8), dim3(256), 0, stream>>>(
      icp, zcbp, w2p, cates, x, delta, b2, W3, b3, out);
}
